// VertebraeCharacteristicsLoss_80788334838315
// MI455X (gfx1250) — compile-verified, matrix operand order checked by reading, not run on hardware
//
#include <hip/hip_runtime.h>
#include <stdint.h>

#define BB 8
#define HH 512
#define WW 512
#define HW   (HH * WW)        // 262144 = 2^18
#define NPIX (BB * HW)        // 2097152

typedef __attribute__((ext_vector_type(16))) _Float16 v16h;
typedef __attribute__((ext_vector_type(8)))  float    v8f;

// mean inter-vertebra distances (MD) from the reference
__constant__ float c_MD[32] = {
  14.f, 14.f, 18.f, 18.f, 18.5f, 19.f, 19.5f, 20.f, 20.f, 20.f,
  20.5f, 21.f, 21.5f, 22.f, 22.5f, 23.f, 24.5f, 24.5f, 26.5f, 28.5f,
  29.5f, 33.f, 33.f, 33.f, 33.f, 33.f, 33.f, 30.f, 30.f, 30.f, 30.f, 30.f };

// ---- workspace layout (bytes) ----
#define OFF_DESC   0        // int
#define OFF_VERT   4        // int
#define OFF_CENTER 8        // float
#define OFF_PMMIN  16       // int[8]  (init INT_MAX)
#define OFF_U      48       // int[8]
#define OFF_LAB    80       // uint[8]
#define OFF_CNT    112      // int[256]
#define OFF_SH     1136     // int[256]
#define OFF_SW     2160     // int[256]
#define OFF_MED    3184     // float[B*W] = 4096
#define OFF_MC     19568    // int[H*W]  = 262144
#define OFF_HASH   1068144  // u64 hash tables, 8-byte aligned

// ---------------------------------------------------------------------------
// WMMA-based wave32 sum: lane value x goes into A[m,0] (lanes 0-15) and
// A[m,8] (lanes 16-31) of a 16x32 f16 A-matrix; B = ones => D[m,n] =
// x[m] + x[m+16]. Summing the 8 C/D VGPRs per lane gives half-sums; one
// cross-half shuffle yields the full wave sum in every lane.
// Exact for integer-valued x <= 2048 (f16 integer range), accum in f32.
// ---------------------------------------------------------------------------
__device__ __forceinline__ float wave_sum_wmma(float x) {
  v16h a = {};
  a[0] = (_Float16)x;
  v16h bm;
#pragma unroll
  for (int i = 0; i < 16; ++i) bm[i] = (_Float16)1.0f;
  v8f c = {};
  c = __builtin_amdgcn_wmma_f32_16x16x32_f16(
      /*neg_a=*/false, a, /*neg_b=*/false, bm,
      /*c_mod=*/(short)0, c, /*reuse_a=*/false, /*reuse_b=*/false);
  float s = c[0] + c[1] + c[2] + c[3] + c[4] + c[5] + c[6] + c[7];
  s += __shfl_xor(s, 16, 32);
  return s;
}

__device__ __forceinline__ void block_atomic_add_int(float x, int* dst, float* wsum) {
  float s = wave_sum_wmma(x);
  int lane = threadIdx.x & 31;
  int wid  = threadIdx.x >> 5;
  if (lane == 0) wsum[wid] = s;
  __syncthreads();
  if (threadIdx.x == 0) {
    float tot = 0.f;
    int nw = (int)((blockDim.x + 31) >> 5);
    for (int i = 0; i < nw; ++i) tot += wsum[i];
    atomicAdd(dst, (int)(tot + 0.5f));
  }
}

// ---------------------------------------------------------------------------
// Async global->LDS copy of one 32-bit word per lane (CDNA5 ASYNCcnt path).
// lds_off = wave-relative LDS byte offset (= low 32 bits of generic pointer,
// per ISA aperture mapping), gaddr = 64-bit global byte address.
// ---------------------------------------------------------------------------
__device__ __forceinline__ void async_copy_b32(unsigned lds_off, const float* gaddr) {
#if defined(__gfx1250__)
  asm volatile("global_load_async_to_lds_b32 %0, %1, off"
               :
               : "v"(lds_off), "v"((unsigned long long)(uintptr_t)gaddr)
               : "memory");
#else
  // fallback: synchronous copy
  *(float*)(size_t)lds_off = *gaddr;  // never used on gfx1250
#endif
}

__device__ __forceinline__ void async_wait0() {
#if defined(__gfx1250__)
  asm volatile("s_wait_asynccnt 0" ::: "memory");
#endif
}

// ---------------------------------------------------------------------------
__global__ void zero_ws_kernel(unsigned* ws, long long nwords) {
  long long i = (long long)blockIdx.x * blockDim.x + threadIdx.x;
  long long stride = (long long)gridDim.x * blockDim.x;
  for (; i < nwords; i += stride) {
    unsigned v = 0u;
    if (i >= 4 && i < 12) v = 0x7FFFFFFFu;   // pmmin[b] = INT_MAX
    ws[i] = v;
  }
}

// ---------------------------------------------------------------------------
// Descending loss: one block per (b,h) row; row staged into LDS with
// GLOBAL_LOAD_ASYNC_TO_LDS_B32 (ASYNCcnt), then rounded in place.
// invalid += sum over s=1..29 of [pm[w] - pred[w+s]*mask[w+s]*mask[w] < 0]
//          + multiplicity term for pm[w] < 0 in the tail (faithful).
// ---------------------------------------------------------------------------
__global__ void desc_kernel(const float* __restrict__ pred,
                            const float* __restrict__ mask,
                            int* g_desc) {
  __shared__ float sp[WW], sm[WW], wsum[8];
  const int row = blockIdx.x;
  const int t = threadIdx.x;
  const float* pr = pred + (size_t)row * WW;
  const float* mr = mask + (size_t)row * WW;

  unsigned sp_off = (unsigned)(uintptr_t)(void*)&sp[0];
  unsigned sm_off = (unsigned)(uintptr_t)(void*)&sm[0];
  async_copy_b32(sp_off + 4u * (unsigned)t,         pr + t);
  async_copy_b32(sp_off + 4u * (unsigned)(t + 256), pr + t + 256);
  async_copy_b32(sm_off + 4u * (unsigned)t,         mr + t);
  async_copy_b32(sm_off + 4u * (unsigned)(t + 256), mr + t + 256);
  async_wait0();                 // this wave's async LDS writes complete
  __syncthreads();               // all waves' tiles visible
  sp[t]       = rintf(sp[t]);
  sp[t + 256] = rintf(sp[t + 256]);
  __syncthreads();

  float cnt = 0.f;
#pragma unroll
  for (int k = 0; k < 2; ++k) {
    int w = t + k * 256;
    float prw = sp[w], mw = sm[w];
    float pmw = prw * mw;
#pragma unroll
    for (int s = 1; s <= 29; ++s) {
      if (w + s < WW) {
        float ps = sp[w + s] * sm[w + s] * mw;
        if (pmw - ps < 0.f) cnt += 1.f;
      }
    }
    if (w >= WW - 29 && pmw < 0.f) cnt += (float)(w - (WW - 30));
  }
  block_atomic_add_int(cnt, g_desc, wsum);
}

// ---------------------------------------------------------------------------
// maskcount[h,w] = #{b : mask[b,h,w] != 0}   (collapses the faithful B x B
// broadcast of the vertical-equal loss)
// ---------------------------------------------------------------------------
__global__ void maskcount_kernel(const float* __restrict__ mask, int* __restrict__ mc) {
  int idx = blockIdx.x * 256 + threadIdx.x;    // 0 .. H*W-1
  int c = 0;
#pragma unroll
  for (int b = 0; b < BB; ++b)
    c += (mask[(size_t)b * HW + idx] != 0.f) ? 1 : 0;
  mc[idx] = c;
}

// ---------------------------------------------------------------------------
// Per-column lower median of the nonzero pm values (ints 1..26) via histogram.
// One thread per (b,w) column; coalesced along w.
// ---------------------------------------------------------------------------
__global__ void median_kernel(const float* __restrict__ pred,
                              const float* __restrict__ mask,
                              float* __restrict__ med) {
  int t = blockIdx.x * 256 + threadIdx.x;      // 0 .. B*W-1
  int b = t >> 9;
  int w = t & (WW - 1);
  const float* pb = pred + (size_t)b * HW + w;
  const float* mb = mask + (size_t)b * HW + w;
  int hist[27];
#pragma unroll
  for (int i = 0; i < 27; ++i) hist[i] = 0;
  for (int h = 0; h < HH; ++h) {
    if ((h & 63) == 0 && h + 64 < HH) {
      __builtin_prefetch(pb + (size_t)(h + 64) * WW, 0, 1);   // global_prefetch_b8
      __builtin_prefetch(mb + (size_t)(h + 64) * WW, 0, 1);
    }
    float v = rintf(pb[(size_t)h * WW]) * mb[(size_t)h * WW];
    int iv = (int)v;
    if (iv >= 1 && iv <= 26) hist[iv]++;
  }
  int n = 0;
#pragma unroll
  for (int i = 1; i < 27; ++i) n += hist[i];
  float m = 0.f;
  if (n > 0) {
    int idx = (n - 1) >> 1;                    // torch lower-median index
    int c = 0;
    for (int v = 1; v < 27; ++v) {
      c += hist[v];
      if (idx < c) { m = (float)v; break; }
    }
  }
  med[t] = m;
}

// ---------------------------------------------------------------------------
// vertical-equal count: sum over (b,h,w) of [round(pred)-med != 0] * maskcount
// ---------------------------------------------------------------------------
__global__ void vert_count_kernel(const float* __restrict__ pred,
                                  const float* __restrict__ med,
                                  const int* __restrict__ mc,
                                  int* g_vert) {
  __shared__ float wsum[8];
  int idx = blockIdx.x * 256 + threadIdx.x;
  int b = idx >> 18;
  int r = idx & (HW - 1);
  int w = r & (WW - 1);
  float diff = rintf(pred[idx]) - med[(b << 9) + w];
  float cnt = (diff != 0.f) ? (float)mc[r] : 0.f;   // <= 8, f16-exact
  block_atomic_add_int(cnt, g_vert, wsum);
}

// ---------------------------------------------------------------------------
// Center-dist pass 1: per-(b,label) integer segment sums via LDS histograms.
// Each 256-thread block covers pixels of a single b (HW/256 = 1024 blocks/b).
// ---------------------------------------------------------------------------
__global__ void center1_kernel(const float* __restrict__ pred,
                               const float* __restrict__ mask,
                               int* g_cnt, int* g_sh, int* g_sw, int* g_min) {
  __shared__ int lc[32], lh[32], lw[32], lmin;
  int t = threadIdx.x;
  if (t < 32) { lc[t] = 0; lh[t] = 0; lw[t] = 0; }
  if (t == 0) lmin = 0x7FFFFFFF;
  __syncthreads();
  int idx = blockIdx.x * 256 + t;
  int b = idx >> 18;
  int r = idx & (HW - 1);
  int h = r >> 9, w = r & (WW - 1);
  float pm = rintf(pred[idx]) * mask[idx];
  int ip = (int)pm;                             // truncation, matches astype(int32)
  int lab = ip < 0 ? 0 : (ip > 31 ? 31 : ip);
  atomicAdd(&lc[lab], 1);
  atomicAdd(&lh[lab], h);
  atomicAdd(&lw[lab], w);
  atomicMin(&lmin, ip);
  __syncthreads();
  if (t < 32 && lc[t]) {
    atomicAdd(&g_cnt[b * 32 + t], lc[t]);
    atomicAdd(&g_sh[b * 32 + t], lh[t]);
    atomicAdd(&g_sw[b * 32 + t], lw[t]);
  }
  if (t == 0) atomicMin(&g_min[b], lmin);
}

__global__ void center2_kernel(const int* g_cnt, const int* g_sh, const int* g_sw,
                               const int* g_min, float* out_center) {
  __shared__ float buf[256];
  int t = threadIdx.x;
  int b = t >> 5, i = t & 31;
  float contrib = 0.f;
  if (i >= 1) {
    int c1 = g_cnt[b * 32 + i], c0 = g_cnt[b * 32 + i - 1];
    if (c1 > 0 && c0 > 0 && i >= g_min[b] + 2) {
      float y1 = (float)g_sh[b * 32 + i]     / (float)c1;
      float x1 = (float)g_sw[b * 32 + i]     / (float)c1;
      float y0 = (float)g_sh[b * 32 + i - 1] / (float)c0;
      float x0 = (float)g_sw[b * 32 + i - 1] / (float)c0;
      float dx = x1 - x0, dy = y1 - y0;
      float dist = sqrtf(dx * dx + dy * dy);
      contrib = fabsf(dist - c_MD[i]);
    }
  }
  buf[t] = contrib;
  __syncthreads();
  if (t == 0) {
    float s = 0.f;
    for (int k = 0; k < 256; ++k) s += buf[k];
    *out_center = s;
  }
}

// ---------------------------------------------------------------------------
// Region-proposal: count distinct (weak-label>0, raw-pred-bits) pairs per
// sample via per-sample open-addressing hash table; label presence bitmask.
// ---------------------------------------------------------------------------
__device__ __forceinline__ unsigned long long mix64(unsigned long long z) {
  z += 0x9E3779B97F4A7C15ull;
  z = (z ^ (z >> 30)) * 0xBF58476D1CE4E5B9ull;
  z = (z ^ (z >> 27)) * 0x94D049BB133111EBull;
  return z ^ (z >> 31);
}

__global__ void region_kernel(const float* __restrict__ pred,
                              const int* __restrict__ weak,
                              int* g_u, unsigned* g_lab,
                              unsigned long long* g_hash, unsigned slots) {
  int idx = blockIdx.x * 256 + threadIdx.x;
  int b = idx >> 18;
  int wv = weak[idx];
  if (wv > 0) {
    atomicOr(&g_lab[b], 1u << (wv & 31));
    if (slots > 0) {
      unsigned long long key =
          ((unsigned long long)(unsigned)wv << 32) | (unsigned long long)__float_as_uint(pred[idx]);
      unsigned long long* tab = g_hash + (size_t)b * slots;
      unsigned m = slots - 1;
      unsigned j = (unsigned)mix64(key) & m;
      for (unsigned p = 0; p < slots; ++p) {
        unsigned long long old = atomicCAS(&tab[j], 0ull, key);
        if (old == 0ull) { atomicAdd(&g_u[b], 1); break; }
        if (old == key) break;
        j = (j + 1) & m;
      }
    }
  }
}

// ---------------------------------------------------------------------------
__global__ void finalize_kernel(const int* g_desc, const int* g_vert,
                                const float* g_center, const int* g_u,
                                const unsigned* g_lab, float* out) {
  if (blockIdx.x == 0 && threadIdx.x == 0) {
    float l = ((float)(*g_desc) / (float)NPIX) * 20.f;
    l += (float)(*g_vert) / (float)NPIX;
    l += (*g_center) / 1000.f;
    float r = 0.f;                              // faithful sequential divide chain
    for (int b = 0; b < BB; ++b) {
      int nl = __popc(g_lab[b]);
      r += (float)(g_u[b] - nl);
      if (nl > 0) r /= (float)nl;
    }
    l += r / 100.f;
    out[0] = l;
  }
}

// ---------------------------------------------------------------------------
extern "C" void kernel_launch(void* const* d_in, const int* in_sizes, int n_in,
                              void* d_out, int out_size, void* d_ws, size_t ws_size,
                              hipStream_t stream) {
  (void)in_sizes; (void)n_in; (void)out_size;
  const float* pred = (const float*)d_in[1];      // predictions [B,1,H,W] f32
  const float* mask = (const float*)d_in[2];      // detection_mask [B,H,W] f32
  const int*   weak = (const int*)d_in[3];        // weak_mask [B,H,W] i32
  float* out = (float*)d_out;
  char* ws = (char*)d_ws;

  int*      g_desc   = (int*)(ws + OFF_DESC);
  int*      g_vert   = (int*)(ws + OFF_VERT);
  float*    g_center = (float*)(ws + OFF_CENTER);
  int*      g_min    = (int*)(ws + OFF_PMMIN);
  int*      g_u      = (int*)(ws + OFF_U);
  unsigned* g_lab    = (unsigned*)(ws + OFF_LAB);
  int*      g_cnt    = (int*)(ws + OFF_CNT);
  int*      g_sh     = (int*)(ws + OFF_SH);
  int*      g_sw     = (int*)(ws + OFF_SW);
  float*    g_med    = (float*)(ws + OFF_MED);
  int*      g_mc     = (int*)(ws + OFF_MC);
  unsigned long long* g_hash = (unsigned long long*)(ws + OFF_HASH);

  // size per-sample hash table from available scratch (power of two)
  unsigned slots = 0;
  if (ws_size > (size_t)OFF_HASH + 4096) {
    size_t per = (ws_size - OFF_HASH) / (8ull * BB);   // entries per sample
    if (per >= 1024) {
      unsigned s = 1024;
      while ((size_t)s * 2 <= per && s < (1u << 20)) s <<= 1;
      slots = s;
    }
  }

  long long nwords = OFF_HASH / 4 + (long long)slots * BB * 2;
  zero_ws_kernel<<<4096, 256, 0, stream>>>((unsigned*)ws, nwords);
  desc_kernel<<<BB * HH, 256, 0, stream>>>(pred, mask, g_desc);
  maskcount_kernel<<<HW / 256, 256, 0, stream>>>(mask, g_mc);
  median_kernel<<<(BB * WW) / 256, 256, 0, stream>>>(pred, mask, g_med);
  vert_count_kernel<<<NPIX / 256, 256, 0, stream>>>(pred, g_med, g_mc, g_vert);
  center1_kernel<<<NPIX / 256, 256, 0, stream>>>(pred, mask, g_cnt, g_sh, g_sw, g_min);
  center2_kernel<<<1, 256, 0, stream>>>(g_cnt, g_sh, g_sw, g_min, g_center);
  region_kernel<<<NPIX / 256, 256, 0, stream>>>(pred, weak, g_u, g_lab, g_hash, slots);
  finalize_kernel<<<1, 64, 0, stream>>>(g_desc, g_vert, g_center, g_u, g_lab, out);
}
